// RandLANet_79774722556534
// MI455X (gfx1250) — matrix operand order repaired and checked by reading, not compiled, not run on hardware
//
#include <hip/hip_runtime.h>
#include <math.h>

#define BATCH 2
#define NPTS  8192
#define KNBR  16

typedef __attribute__((ext_vector_type(16))) _Float16 v16h;
typedef __attribute__((ext_vector_type(8)))  _Float16 h8;
typedef __attribute__((ext_vector_type(8)))  float    v8f;
typedef __attribute__((ext_vector_type(2)))  float    f2;

// ---------------------------------------------------------------------------
// Weight pre-pack: W [Co x Ci] f32 -> Wp [coP x ciP] f16, zero padded.
// ---------------------------------------------------------------------------
__global__ void k_packW(const float* __restrict__ W, _Float16* __restrict__ Wp,
                        int Ci, int ciP, int Co, int total)
{
    int t = blockIdx.x * blockDim.x + threadIdx.x;
    if (t >= total) return;
    int r = t / ciP, c = t - r * ciP;
    float v = (r < Co && c < Ci) ? W[(size_t)r * Ci + c] : 0.0f;
    Wp[t] = (_Float16)v;
}

// ---- WMMA fragment loaders ------------------------------------------------
// A (16x32 f16): lanes 0-15 rows, K={0..7,16..23}; lanes 16-31 K={8..15,24..31}
__device__ __forceinline__ v16h loadA(const float* __restrict__ xrow, int k0, int half)
{
    const float* pa0 = xrow + k0 + half * 8;
    const float* pa1 = pa0 + 16;
    v16h a;
    #pragma unroll
    for (int e = 0; e < 4; ++e) {
        f2 v0 = *(const f2*)(pa0 + 2 * e);
        f2 v1 = *(const f2*)(pa1 + 2 * e);
        a[2 * e]     = (_Float16)v0[0];
        a[2 * e + 1] = (_Float16)v0[1];
        a[8 + 2 * e] = (_Float16)v1[0];
        a[9 + 2 * e] = (_Float16)v1[1];
    }
    return a;
}
// B (32x16 f16): lanes 0-15 cols K={0..7,16..23}; lanes 16-31 K={8..15,24..31} (packed f16)
__device__ __forceinline__ v16h loadB(const _Float16* __restrict__ wrow, int k0, int half)
{
    h8 lo = *(const h8*)(wrow + k0 + half * 8);
    h8 hi = *(const h8*)(wrow + k0 + half * 8 + 16);
    v16h b;
    #pragma unroll
    for (int e = 0; e < 8; ++e) { b[e] = lo[e]; b[8 + e] = hi[e]; }
    return b;
}

// ---------------------------------------------------------------------------
// Generic fused linear layer:  Y = act( bn( X @ W^T + b ) )
// X: [M x Ci] f32 (ldx even, M % 16 == 0, buffer has >=256B slack),
// Wp: packed f16 [ceil16(Co) x ciP], zero padded.
// One wave per 16x16 tile; K blocked by 32; main loop unrolled x2 with two
// independent accumulator chains (avoids WMMA->VALU WAR hazard stalls).
// act: 0=none, 1=relu, 2=leaky(slope)
// ---------------------------------------------------------------------------
__global__ __launch_bounds__(32) void k_linear(
    const float* __restrict__ X, int ldx,
    const _Float16* __restrict__ Wp, int ciP,
    const float* __restrict__ bias,
    const float* __restrict__ g, const float* __restrict__ beta, int use_bn,
    float* __restrict__ Y, int ldy, int yoff,
    int M, int Ci, int Co, int act, float slope)
{
    const int m0   = blockIdx.x * 16;
    const int n0   = blockIdx.y * 16;
    const int lane = threadIdx.x;      // 0..31
    const int half = lane >> 4;        // 0 or 1
    const int lr   = lane & 15;        // A-row / B-col within tile

    v8f acc0 = {}, acc1 = {};

    const float*    xrow = X  + (size_t)(m0 + lr) * ldx;   // M mult of 16 => safe
    const _Float16* wrow = Wp + (size_t)(n0 + lr) * ciP;   // rows padded => safe

    const int kEnd = Ci & ~31;         // full 32-blocks
    int k0 = 0;
    // main loop: 2 blocks/iter, independent fragments + accumulators
    for (; k0 + 64 <= kEnd; k0 += 64) {
        v16h a0 = loadA(xrow, k0,      half);
        v16h b0 = loadB(wrow, k0,      half);
        v16h a1 = loadA(xrow, k0 + 32, half);
        v16h b1 = loadB(wrow, k0 + 32, half);
        acc0 = __builtin_amdgcn_wmma_f32_16x16x32_f16(false, a0, false, b0,
                                                      (short)0, acc0, false, false);
        acc1 = __builtin_amdgcn_wmma_f32_16x16x32_f16(false, a1, false, b1,
                                                      (short)0, acc1, false, false);
    }
    if (k0 < kEnd) {                   // one remaining full block
        v16h a0 = loadA(xrow, k0, half);
        v16h b0 = loadB(wrow, k0, half);
        acc0 = __builtin_amdgcn_wmma_f32_16x16x32_f16(false, a0, false, b0,
                                                      (short)0, acc0, false, false);
        k0 += 32;
    }
    if (k0 < Ci) {                     // masked K-tail (Ci in {8,10,16} only)
        v16h a;
        #pragma unroll
        for (int e = 0; e < 16; ++e) {
            int k = k0 + ((e < 8) ? (half * 8 + e) : (16 + half * 8 + (e - 8)));
            float v = xrow[k];         // unconditional; slack-padded buffer
            a[e] = (_Float16)((k < Ci) ? v : 0.0f);
        }
        v16h b = loadB(wrow, k0, half);   // zero-padded weights
        acc1 = __builtin_amdgcn_wmma_f32_16x16x32_f16(false, a, false, b,
                                                      (short)0, acc1, false, false);
    }
    #pragma unroll
    for (int e = 0; e < 8; ++e) acc0[e] += acc1[e];

    // ---- epilogue: C layout lanes0-15 -> rows m0+0..7, lanes16-31 -> m0+8..15
    const int bcol = n0 + lr;
    if (bcol < Co) {
        const float bsv = bias ? bias[bcol] : 0.0f;
        float gs = 1.0f, bb = 0.0f;
        if (use_bn) { gs = g[bcol] * rsqrtf(1.0f + 1e-6f); bb = beta[bcol]; }
        #pragma unroll
        for (int v = 0; v < 8; ++v) {
            int m = m0 + half * 8 + v;
            float y = acc0[v] + bsv;
            if (use_bn) y = y * gs + bb;
            if (act == 1)      y = fmaxf(y, 0.0f);
            else if (act == 2) y = (y > 0.0f) ? y : y * slope;
            Y[(size_t)m * ldy + yoff + bcol] = y;
        }
    }
}

// ---------------------------------------------------------------------------
// fc_start (6->8) fused with bn_start + leaky(0.2). Tiny: pure VALU.
// ---------------------------------------------------------------------------
__global__ void k_fc_start(const float* __restrict__ inp,
                           const float* __restrict__ w, const float* __restrict__ b,
                           const float* __restrict__ g, const float* __restrict__ beta,
                           float* __restrict__ out, int total)
{
    int t = blockIdx.x * blockDim.x + threadIdx.x;
    if (t >= total) return;
    const float* x = inp + (size_t)t * 6;
    float xv[6];
    #pragma unroll
    for (int i = 0; i < 6; ++i) xv[i] = x[i];
    float rs = rsqrtf(1.0f + 1e-6f);
    #pragma unroll
    for (int o = 0; o < 8; ++o) {
        float acc = b[o];
        #pragma unroll
        for (int i = 0; i < 6; ++i) acc = fmaf(w[o * 6 + i], xv[i], acc);
        float y = acc * (g[o] * rs) + beta[o];
        out[(size_t)t * 8 + o] = (y > 0.0f) ? y : 0.2f * y;
    }
}

// ---------------------------------------------------------------------------
// Brute-force KNN (K=16) among first Nr points of each batch; coords stride 6.
// One thread/query; 128-pt ref tiles staged in LDS; prefetch next tile;
// 16-deep insertion sort, strict '<' => stable tie order like top_k.
// Requires Nr % 128 == 0 (8192/2048/512/128).
// ---------------------------------------------------------------------------
__global__ __launch_bounds__(128) void k_knn(const float* __restrict__ pts, int Nr,
                                             int* __restrict__ idxO,
                                             float* __restrict__ distO)
{
    __shared__ float sx[128], sy[128], sz[128];
    int b = (int)((blockIdx.x * 128u) / (unsigned)Nr);
    int q = blockIdx.x * 128 + threadIdx.x - b * Nr;
    const float* pq = pts + (size_t)(b * NPTS + q) * 6;
    float qx = pq[0], qy = pq[1], qz = pq[2];

    float bd[KNBR]; int bi[KNBR];
    #pragma unroll
    for (int i = 0; i < KNBR; ++i) { bd[i] = 3.4e38f; bi[i] = 0; }

    for (int c0 = 0; c0 < Nr; c0 += 128) {
        __syncthreads();
        {
            int r = c0 + threadIdx.x;
            const float* p = pts + (size_t)(b * NPTS + r) * 6;
            sx[threadIdx.x] = p[0]; sy[threadIdx.x] = p[1]; sz[threadIdx.x] = p[2];
            int r2 = r + 128;
            if (r2 < Nr) __builtin_prefetch(pts + (size_t)(b * NPTS + r2) * 6, 0, 1);
        }
        __syncthreads();
        for (int j = 0; j < 128; ++j) {
            float dx = qx - sx[j], dy = qy - sy[j], dz = qz - sz[j];
            float d2 = fmaf(dx, dx, fmaf(dy, dy, dz * dz));
            if (d2 < bd[KNBR - 1]) {
                bd[KNBR - 1] = d2; bi[KNBR - 1] = c0 + j;
                #pragma unroll
                for (int s = KNBR - 1; s > 0; --s) {
                    if (bd[s] < bd[s - 1]) {
                        float td = bd[s]; bd[s] = bd[s - 1]; bd[s - 1] = td;
                        int   ti = bi[s]; bi[s] = bi[s - 1]; bi[s - 1] = ti;
                    } else break;
                }
            }
        }
    }
    size_t o = (size_t)(b * Nr + q) * KNBR;
    #pragma unroll
    for (int i = 0; i < KNBR; ++i) {
        idxO[o + i]  = bi[i];
        distO[o + i] = sqrtf(fmaxf(bd[i], 0.0f));
    }
}

// ---------------------------------------------------------------------------
// 1-NN (decoder upsampling): nearest coarse ref index for each fine query.
// ---------------------------------------------------------------------------
__global__ __launch_bounds__(128) void k_nn1(const float* __restrict__ pts,
                                             int Nref, int Nq, int* __restrict__ idxO)
{
    __shared__ float sx[128], sy[128], sz[128];
    int b = (int)((blockIdx.x * 128u) / (unsigned)Nq);
    int q = blockIdx.x * 128 + threadIdx.x - b * Nq;
    const float* pq = pts + (size_t)(b * NPTS + q) * 6;
    float qx = pq[0], qy = pq[1], qz = pq[2];
    float best = 3.4e38f; int bidx = 0;
    for (int c0 = 0; c0 < Nref; c0 += 128) {
        __syncthreads();
        int r = c0 + threadIdx.x;
        if (r < Nref) {
            const float* p = pts + (size_t)(b * NPTS + r) * 6;
            sx[threadIdx.x] = p[0]; sy[threadIdx.x] = p[1]; sz[threadIdx.x] = p[2];
        }
        __syncthreads();
        int lim = (Nref - c0 < 128) ? (Nref - c0) : 128;
        for (int j = 0; j < lim; ++j) {
            float dx = qx - sx[j], dy = qy - sy[j], dz = qz - sz[j];
            float d2 = fmaf(dx, dx, fmaf(dy, dy, dz * dz));
            if (d2 < best) { best = d2; bidx = c0 + j; }
        }
    }
    idxO[b * Nq + q] = bidx;
}

// ---------------------------------------------------------------------------
// LSE relative-position features: [q, nb, q-nb, dist] -> 10 channels.
// Depends only on coords/idx/dist => computed once per level, reused by lse2.
// ---------------------------------------------------------------------------
__global__ void k_lse_cat(const float* __restrict__ pts, int Nr,
                          const int* __restrict__ idx, const float* __restrict__ dist,
                          float* __restrict__ cat, int totalRK)
{
    int t = blockIdx.x * blockDim.x + threadIdx.x;
    if (t >= totalRK) return;
    int row = t / KNBR, k = t - row * KNBR;
    int b = row / Nr,   n = row - b * Nr;
    int j = idx[(size_t)row * KNBR + k];
    const float* pq = pts + (size_t)(b * NPTS + n) * 6;
    const float* pr = pts + (size_t)(b * NPTS + j) * 6;
    float* o = cat + (size_t)t * 10;
    float qx = pq[0], qy = pq[1], qz = pq[2];
    float rx = pr[0], ry = pr[1], rz = pr[2];
    o[0] = qx; o[1] = qy; o[2] = qz;
    o[3] = rx; o[4] = ry; o[5] = rz;
    o[6] = qx - rx; o[7] = qy - ry; o[8] = qz - rz;
    o[9] = dist[(size_t)row * KNBR + k];
}

// Broadcast per-point feats into channels [off, off+dh) of the (rows,K,ldo) buf.
__global__ void k_bcast(const float* __restrict__ f, float* __restrict__ out,
                        int dh, int ldo, int off, int total)
{
    int t = blockIdx.x * blockDim.x + threadIdx.x;
    if (t >= total) return;
    int row = t / (KNBR * dh), rem = t - row * (KNBR * dh);
    int k = rem / dh, c = rem - k * dh;
    out[((size_t)row * KNBR + k) * ldo + off + c] = f[(size_t)row * dh + c];
}

// Attention pool: per (row, channel) softmax over K=16 scores, weighted sum.
__global__ void k_attpool(const float* __restrict__ scores, const float* __restrict__ x,
                          float* __restrict__ out, int d, int total)
{
    int t = blockIdx.x * blockDim.x + threadIdx.x;
    if (t >= total) return;
    int row = t / d, c = t - row * d;
    size_t base = (size_t)row * KNBR * d + c;
    float m = -3.4e38f;
    #pragma unroll
    for (int k = 0; k < KNBR; ++k) m = fmaxf(m, scores[base + (size_t)k * d]);
    float ssum = 0.0f, acc = 0.0f;
    #pragma unroll
    for (int k = 0; k < KNBR; ++k) {
        float e = __expf(scores[base + (size_t)k * d] - m);
        ssum += e;
        acc  = fmaf(e, x[base + (size_t)k * d], acc);
    }
    out[(size_t)row * d + c] = acc / ssum;
}

__global__ void k_add_leaky(const float* __restrict__ a, const float* __restrict__ b,
                            float* __restrict__ o, float slope, int total)
{
    int t = blockIdx.x * blockDim.x + threadIdx.x;
    if (t >= total) return;
    float y = a[t] + b[t];
    o[t] = (y > 0.0f) ? y : y * slope;
}

// Compact prefix subsample: out(b, n<Nout, c) = in(b*Nin + n, c)
__global__ void k_subsample(const float* __restrict__ in, int Nin,
                            float* __restrict__ out, int Nout, int C, int total)
{
    int t = blockIdx.x * blockDim.x + threadIdx.x;
    if (t >= total) return;
    int b = t / (Nout * C), rem = t - b * (Nout * C);
    int n = rem / C, c = rem - n * C;
    out[t] = in[((size_t)b * Nin + n) * C + c];
}

// Decoder: out(b,n,0:Cx) = x[idx1], out(b,n,Cx:Cx+Cs) = skip
__global__ void k_gather_concat(const float* __restrict__ x, int Nref, int Cx,
                                const int* __restrict__ idx1,
                                const float* __restrict__ skip, int Cs,
                                float* __restrict__ out, int Nq, int total)
{
    int t = blockIdx.x * blockDim.x + threadIdx.x;
    if (t >= total) return;
    int Ct = Cx + Cs;
    int b = t / (Nq * Ct), rem = t - b * (Nq * Ct);
    int n = rem / Ct, c = rem - n * Ct;
    int row = b * Nq + n;
    float v;
    if (c < Cx) v = x[((size_t)b * Nref + idx1[row]) * Cx + c];
    else        v = skip[(size_t)row * Cs + (c - Cx)];
    out[t] = v;
}

// (B,N,3) -> (B,3,N)
__global__ void k_transpose(const float* __restrict__ y, float* __restrict__ out, int total)
{
    int t = blockIdx.x * blockDim.x + threadIdx.x;
    if (t >= total) return;
    int b = t / (3 * NPTS), rem = t - b * (3 * NPTS);
    int c = rem / NPTS, n = rem - c * NPTS;
    out[t] = y[((size_t)b * NPTS + n) * 3 + c];
}

// ===========================================================================
extern "C" void kernel_launch(void* const* d_in, const int* in_sizes, int n_in,
                              void* d_out, int out_size, void* d_ws, size_t ws_size,
                              hipStream_t stream)
{
    (void)in_sizes; (void)n_in; (void)out_size; (void)ws_size;

    // ---- workspace bump allocator (256B slack per buffer for safe overread)
    char* base = (char*)d_ws;
    size_t off = 0;
    auto allocf = [&](size_t n) { float*    p = (float*)   (base + off); off += ((n * 4 + 255) / 256) * 256 + 256; return p; };
    auto alloci = [&](size_t n) { int*      p = (int*)     (base + off); off += ((n * 4 + 255) / 256) * 256 + 256; return p; };
    auto alloch = [&](size_t n) { _Float16* p = (_Float16*)(base + off); off += ((n * 2 + 255) / 256) * 256 + 256; return p; };

    // ---- parameter walk: setup_inputs() dict-insertion order, with packing
    int ii = 0;
    const float* input = (const float*)d_in[ii++];
    auto nf = [&]() { return (const float*)d_in[ii++]; };

    struct PW { const _Float16* w; int ciP; };
    auto packW = [&](const float* w, int Ci, int Co) -> PW {
        int ciP = (Ci + 31) & ~31;
        int coP = (Co + 15) & ~15;
        _Float16* p = alloch((size_t)coP * ciP);
        int total = coP * ciP;
        k_packW<<<(total + 255) / 256, 256, 0, stream>>>(w, p, Ci, ciP, Co, total);
        return {p, ciP};
    };

    struct MLPp { PW w; const float *b, *g, *beta; };
    auto mlp = [&](int Ci, int Co, bool bn) {
        MLPp m;
        m.w = packW(nf(), Ci, Co);
        m.b = nf();
        m.g = bn ? nf() : nullptr; m.beta = bn ? nf() : nullptr;
        return m;
    };
    struct PoolP { PW score; MLPp m; };
    auto pool = [&](int Ci, int Co) { PoolP p; p.score = packW(nf(), Ci, Ci); p.m = mlp(Ci, Co, true); return p; };
    struct LfaP { MLPp mlp1, lse1; PoolP pool1; MLPp lse2; PoolP pool2; MLPp mlp2, shortp; };

    const int NL[4] = {8192, 2048, 512, 128};
    const int DI[4] = {8, 32, 128, 256};
    const int DO[4] = {16, 64, 128, 256};

    const float* fcs_w = nf(); const float* fcs_b = nf();      // fc_start (raw f32)
    const float* bn_g = nf();  const float* bn_beta = nf();    // bn_start
    LfaP enc[4];
    for (int l = 0; l < 4; ++l) {
        int d = DO[l], dh = d / 2;
        enc[l].mlp1   = mlp(DI[l], dh, false);
        enc[l].lse1   = mlp(10, dh, true);
        enc[l].pool1  = pool(d, dh);
        enc[l].lse2   = mlp(10, dh, true);
        enc[l].pool2  = pool(d, d);
        enc[l].mlp2   = mlp(d, 2 * d, false);
        enc[l].shortp = mlp(DI[l], 2 * d, true);
    }
    MLPp mid = mlp(512, 512, false);
    const int CTd[4] = {1024, 512, 256, 64};
    const int COo[4] = {256, 128, 32, 8};
    MLPp dec[4];
    for (int j = 0; j < 4; ++j) dec[j] = mlp(CTd[j], COo[j], true);
    MLPp fc1 = mlp(8, 64, true), fc2 = mlp(64, 32, true), fc3 = mlp(32, 3, false);

    // ---- activation / index buffers --------------------------------------
    float* x0 = allocf((size_t)BATCH * NPTS * 8);
    int* idxL[4]; float* distL[4];
    for (int l = 0; l < 4; ++l) {
        idxL[l]  = alloci((size_t)BATCH * NL[l] * KNBR);
        distL[l] = allocf((size_t)BATCH * NL[l] * KNBR);
    }
    float* cat    = allocf((size_t)BATCH * 8192 * KNBR * 10);
    float* big0   = allocf(4194304);   // (B,N,K,d) max
    float* big1   = allocf(4194304);   // scores
    float* bx1    = allocf(131072);    // (B,N,d/2)
    float* pooled = allocf(262144);    // (B,N,d)
    float* bp2    = allocf(262144);
    float* bm     = allocf(524288);
    float* bsh    = allocf(524288);
    float* fin2   = allocf(131072);    // subsampled lfa input
    float* stk[4];
    for (int l = 0; l < 4; ++l) stk[l] = allocf((size_t)BATCH * NL[l] * 2 * DO[l]);
    float* midIn  = allocf((size_t)BATCH * 32 * 512);
    float* midOut = allocf((size_t)BATCH * 32 * 512);
    int*   idx1   = alloci((size_t)BATCH * 8192);
    float* decA   = allocf(1048576);
    float* decX0  = allocf(1048576);
    float* decX1  = allocf(1048576);

    // ---- launch helpers ---------------------------------------------------
    auto linear = [&](const float* X, int ldx, const MLPp& p, int bn,
                      float* Y, int ldy, int yoff, int M, int Ci, int Co,
                      int act, float slope) {
        dim3 g((M + 15) / 16, (Co + 15) / 16);
        k_linear<<<g, 32, 0, stream>>>(X, ldx, p.w.w, p.w.ciP, p.b, p.g, p.beta, bn,
                                       Y, ldy, yoff, M, Ci, Co, act, slope);
    };
    auto linS = [&](const float* X, int ldx, const PW& W, float* Y, int ldy,
                    int M, int C) {   // score GEMM: no bias/bn/act
        dim3 g((M + 15) / 16, (C + 15) / 16);
        k_linear<<<g, 32, 0, stream>>>(X, ldx, W.w, W.ciP, nullptr, nullptr, nullptr, 0,
                                       Y, ldy, 0, M, C, C, 0, 0.0f);
    };
    auto gridOf = [](size_t total) { return (unsigned)((total + 255) / 256); };

    // ---- stem + all KNNs --------------------------------------------------
    k_fc_start<<<gridOf((size_t)BATCH * NPTS), 256, 0, stream>>>(
        input, fcs_w, fcs_b, bn_g, bn_beta, x0, BATCH * NPTS);
    for (int l = 0; l < 4; ++l)
        k_knn<<<(BATCH * NL[l]) / 128, 128, 0, stream>>>(input, NL[l], idxL[l], distL[l]);

    // ---- encoder ----------------------------------------------------------
    for (int l = 0; l < 4; ++l) {
        const int Nr = NL[l], d = DO[l], dh = d / 2;
        const int rows = BATCH * Nr, rowsK = rows * KNBR;
        const float* fin = x0;
        if (l > 0) {
            k_subsample<<<gridOf((size_t)BATCH * Nr * DI[l]), 256, 0, stream>>>(
                stk[l - 1], NL[l - 1], fin2, Nr, DI[l], BATCH * Nr * DI[l]);
            fin = fin2;
        }
        // mlp1: din -> d/2, leaky 0.2
        linear(fin, DI[l], enc[l].mlp1, 0, bx1, dh, 0, rows, DI[l], dh, 2, 0.2f);
        // relative-position encoding (shared by lse1/lse2)
        k_lse_cat<<<gridOf((size_t)rowsK), 256, 0, stream>>>(
            input, Nr, idxL[l], distL[l], cat, rowsK);
        // lse1: 10 -> d/2 (bn, relu) into channels [0,dh); broadcast feats into [dh,d)
        linear(cat, 10, enc[l].lse1, 1, big0, d, 0, rowsK, 10, dh, 1, 0.0f);
        k_bcast<<<gridOf((size_t)rowsK * dh), 256, 0, stream>>>(
            bx1, big0, dh, d, dh, rowsK * dh);
        // pool1
        linS(big0, d, enc[l].pool1.score, big1, d, rowsK, d);
        k_attpool<<<gridOf((size_t)rows * d), 256, 0, stream>>>(big1, big0, pooled, d, rows * d);
        linear(pooled, d, enc[l].pool1.m, 1, bx1, dh, 0, rows, d, dh, 1, 0.0f);
        // lse2 (reuses cat)
        linear(cat, 10, enc[l].lse2, 1, big0, d, 0, rowsK, 10, dh, 1, 0.0f);
        k_bcast<<<gridOf((size_t)rowsK * dh), 256, 0, stream>>>(
            bx1, big0, dh, d, dh, rowsK * dh);
        // pool2
        linS(big0, d, enc[l].pool2.score, big1, d, rowsK, d);
        k_attpool<<<gridOf((size_t)rows * d), 256, 0, stream>>>(big1, big0, pooled, d, rows * d);
        linear(pooled, d, enc[l].pool2.m, 1, bp2, d, 0, rows, d, d, 1, 0.0f);
        // residual: mlp2(x) + short(feats), leaky 0.01
        linear(bp2, d, enc[l].mlp2, 0, bm, 2 * d, 0, rows, d, 2 * d, 0, 0.0f);
        linear(fin, DI[l], enc[l].shortp, 1, bsh, 2 * d, 0, rows, DI[l], 2 * d, 0, 0.0f);
        k_add_leaky<<<gridOf((size_t)rows * 2 * d), 256, 0, stream>>>(
            bm, bsh, stk[l], 0.01f, rows * 2 * d);
    }

    // ---- mid --------------------------------------------------------------
    k_subsample<<<gridOf((size_t)BATCH * 32 * 512), 256, 0, stream>>>(
        stk[3], 128, midIn, 32, 512, BATCH * 32 * 512);
    linear(midIn, 512, mid, 0, midOut, 512, 0, BATCH * 32, 512, 512, 1, 0.0f);

    // ---- decoder ----------------------------------------------------------
    const int NRref[4] = {32, 128, 512, 2048};
    const int NQ[4]    = {128, 512, 2048, 8192};
    const int CX[4]    = {512, 256, 128, 32};
    const int CS[4]    = {512, 256, 128, 32};
    const float* xc = midOut;
    for (int j = 0; j < 4; ++j) {
        k_nn1<<<(BATCH * NQ[j]) / 128, 128, 0, stream>>>(input, NRref[j], NQ[j], idx1);
        int Ct = CX[j] + CS[j];
        k_gather_concat<<<gridOf((size_t)BATCH * NQ[j] * Ct), 256, 0, stream>>>(
            xc, NRref[j], CX[j], idx1, stk[3 - j], CS[j], decA, NQ[j],
            BATCH * NQ[j] * Ct);
        float* outb = (j & 1) ? decX1 : decX0;
        linear(decA, Ct, dec[j], 1, outb, COo[j], 0, BATCH * NQ[j], Ct, COo[j], 1, 0.0f);
        xc = outb;
    }

    // ---- head -------------------------------------------------------------
    linear(decX1, 8, fc1, 1, decX0, 64, 0, BATCH * NPTS, 8, 64, 1, 0.0f);
    linear(decX0, 64, fc2, 1, decA, 32, 0, BATCH * NPTS, 64, 32, 1, 0.0f);
    linear(decA, 32, fc3, 0, decX1, 3, 0, BATCH * NPTS, 32, 3, 0, 0.0f);
    k_transpose<<<gridOf((size_t)BATCH * 3 * NPTS), 256, 0, stream>>>(
        decX1, (float*)d_out, BATCH * 3 * NPTS);
}
